// MutilHeadAttention_72670846648590
// MI455X (gfx1250) — matrix operand order repaired and checked by reading, not compiled, not run on hardware
//
#include <hip/hip_runtime.h>
#include <math.h>

#define NTOK 49
#define NPAD 64
#define CDIM 256
#define HEADS 8
#define HD 32

typedef __attribute__((ext_vector_type(16))) __bf16    v16bf;
typedef __attribute__((ext_vector_type(2)))  __bf16    v2bf;
typedef __attribute__((ext_vector_type(8)))  float     v8f;
typedef __attribute__((ext_vector_type(8)))  unsigned  v8u;

// ---- bf16 helpers: native casts -> v_cvt_pk_bf16_f32 ----
static __device__ __forceinline__ unsigned short bf16s(float f) {
    __bf16 h = (__bf16)f;
    unsigned short s;
    __builtin_memcpy(&s, &h, 2);
    return s;
}
static __device__ __forceinline__ unsigned bf16pack(float lo, float hi) {
    v2bf p;
    p[0] = (__bf16)lo;
    p[1] = (__bf16)hi;
    return __builtin_bit_cast(unsigned, p);
}

// ---- WMMA fragment loaders (wave32, 16-bit data; contiguous k = 16h + s used in both A and B) ----
// A (16x32, MxK): lane L<16 -> row M=L, K = k0+0..15;
//                 lane L>=16 -> row M=L-16, K = k0+16..31.
static __device__ __forceinline__ v16bf load_a_lds(const unsigned short* base,
                                                   int row, int stride, int k0, int hi) {
    int kb = k0 + hi * 16;
    const unsigned* p0 = (const unsigned*)(base + row * stride + kb);
    const unsigned* p1 = (const unsigned*)(base + row * stride + kb + 8);
    v8u u;
    u[0] = p0[0]; u[1] = p0[1]; u[2] = p0[2]; u[3] = p0[3];
    u[4] = p1[0]; u[5] = p1[1]; u[6] = p1[2]; u[7] = p1[3];
    return __builtin_bit_cast(v16bf, u);
}
// B (32x16, KxN): lane L<16 -> col N=L, K=k0..k0+15; lane L>=16 -> col N=L-16, K=k0+16..31.
// base laid out with k contiguous: element (k,n) at base[n*stride + k].
static __device__ __forceinline__ v16bf load_b_lds(const unsigned short* base,
                                                   int col, int stride, int k0, int hi) {
    int kb = k0 + hi * 16;
    const unsigned* p = (const unsigned*)(base + col * stride + kb);
    v8u u;
#pragma unroll
    for (int i = 0; i < 8; ++i) u[i] = p[i];
    return __builtin_bit_cast(v16bf, u);
}
// B fragment from a pre-converted bf16 weight row in global memory (L2-resident).
static __device__ __forceinline__ v16bf load_b_g16(const unsigned short* rowp,
                                                   int k0, int hi) {
    int kb = k0 + hi * 16;
    const unsigned* p = (const unsigned*)(rowp + kb);
    v8u u;
#pragma unroll
    for (int i = 0; i < 8; ++i) u[i] = p[i];
    return __builtin_bit_cast(v16bf, u);
}

static __device__ __forceinline__ v8f wmma_bf16(v16bf a, v16bf b, v8f c) {
    return __builtin_amdgcn_wmma_f32_16x16x32_bf16(false, a, false, b, (short)0, c,
                                                   false, false);
}

// rel-position index: idx[m][n] = c[m] + c[48-n], c[t] = (t/7)*13 + t%7
static __device__ __forceinline__ int rel_idx(int m, int n) {
    int n2 = (NTOK - 1) - n;
    return (m / 7) * 13 + (m % 7) + (n2 / 7) * 13 + (n2 % 7);
}

// =============== prep: bf16 weights + pre-gathered bias (x log2e) ===============
__global__ __launch_bounds__(256)
void swin_prep(const float* __restrict__ w_qkv, const float* __restrict__ w_proj,
               const float* __restrict__ bias_table,
               unsigned short* __restrict__ wqkv_bf,
               unsigned short* __restrict__ wproj_bf,
               float* __restrict__ bias_g) {
    const float LOG2E = 1.4426950408889634f;
    int i = blockIdx.x * 256 + threadIdx.x;
    if (i < 768 * CDIM) wqkv_bf[i] = bf16s(w_qkv[i]);
    if (i < CDIM * CDIM) wproj_bf[i] = bf16s(w_proj[i]);
    if (i < NTOK * NTOK * HEADS) {
        int h = i & 7, mn = i >> 3;
        int m = mn / NTOK, n = mn % NTOK;
        bias_g[i] = bias_table[rel_idx(m, n) * HEADS + h] * LOG2E;  // pre-fold log2e
    }
}

// =============== fused per-window attention =====================================
__global__ __launch_bounds__(256, 1)
void swin_window_attn(const float* __restrict__ x,
                      const unsigned short* __restrict__ wqkv_bf,
                      const float* __restrict__ b_qkv,
                      const unsigned short* __restrict__ wproj_bf,
                      const float* __restrict__ b_proj,
                      const float* __restrict__ bias_g,
                      float* __restrict__ out) {
    extern __shared__ unsigned short smem[];
    unsigned short* xs  = smem;                   // [64][256] x (bf16); reused as O
    unsigned short* qs  = smem + 1 * NPAD * CDIM; // [64][256] q row-major
    unsigned short* ksl = smem + 2 * NPAD * CDIM; // [64][256] k row-major
    unsigned short* vT  = smem + 3 * NPAD * CDIM; // [256][64] v transposed
    unsigned short* pb  = smem + 4 * NPAD * CDIM; // [8][64][64] per-wave probs
    unsigned short* os  = xs;                     // alias: O [64][256]

    const int b    = blockIdx.x;
    const int tid  = threadIdx.x;
    const int lane = tid & 31;
    const int w    = tid >> 5;          // wave id 0..7 == head id
    const int cn   = lane & 15;
    const int hi   = (lane >> 4) & 1;

    // ---------------- phase 0: stage x -> bf16 LDS (float4 -> b64 stores) --------
    const float* xg = x + (size_t)b * NTOK * CDIM;
    for (int i = tid; i < NPAD * CDIM / 4; i += 256) {   // i indexes 4-elem groups
        int r = i >> 6, c4 = (i & 63) * 4;
        uint2 d;
        if (r < NTOK) {
            const float4 f = *(const float4*)(xg + r * CDIM + c4);
            d.x = bf16pack(f.x, f.y);
            d.y = bf16pack(f.z, f.w);
        } else {
            d.x = 0u; d.y = 0u;
        }
        *(uint2*)(xs + (size_t)i * 4) = d;
    }
    __syncthreads();

    // ---------------- phase 1: qkv = x @ w_qkv^T + b_qkv -------------------------
    // 4 M-accumulators live at once: each B (weight) fragment feeds 4 WMMAs.
    for (int nt = w * 6; nt < w * 6 + 6; ++nt) {
        int n = nt * 16 + cn;                  // output feature 0..767
        v8f acc[4] = {};
#pragma unroll
        for (int ks = 0; ks < 8; ++ks) {
            v16bf bm = load_b_g16(wqkv_bf + (size_t)n * CDIM, ks * 32, hi);
#pragma unroll
            for (int mt = 0; mt < 4; ++mt) {
                v16bf a = load_a_lds(xs, mt * 16 + cn, CDIM, ks * 32, hi);
                acc[mt] = wmma_bf16(a, bm, acc[mt]);
            }
        }
        float bias = b_qkv[n];
#pragma unroll
        for (int mt = 0; mt < 4; ++mt)
#pragma unroll
            for (int r = 0; r < 8; ++r) {
                int m = mt * 16 + r + hi * 8;
                unsigned short h16 = bf16s(acc[mt][r] + bias);
                if (n < 256)       qs[m * CDIM + n]          = h16;
                else if (n < 512)  ksl[m * CDIM + (n - 256)] = h16;
                else               vT[(n - 512) * NPAD + m]  = h16;  // transposed v
            }
    }
    __syncthreads();

    // ---------------- phase 2: per-head attention (one wave per head) ------------
    const int h = w;
    const float scale2 = 0.2550347805f;  // 32^-0.5 * log2(e)  (exp2-folded)

    v16bf aF[4], bF[4];
#pragma unroll
    for (int mt = 0; mt < 4; ++mt) aF[mt] = load_a_lds(qs, mt * 16 + cn, CDIM, h * HD, hi);
#pragma unroll
    for (int nt = 0; nt < 4; ++nt) bF[nt] = load_b_lds(ksl, nt * 16 + cn, CDIM, h * HD, hi);

    v8f accS[4][4];
#pragma unroll
    for (int mt = 0; mt < 4; ++mt)
#pragma unroll
        for (int nt = 0; nt < 4; ++nt) {
            v8f z = {};
            accS[mt][nt] = wmma_bf16(aF[mt], bF[nt], z);
        }

    // softmax (base-2, log2e pre-folded into scale & bias); write P (bf16)
    unsigned short* pw = pb + w * NPAD * NPAD;
#pragma unroll
    for (int mt = 0; mt < 4; ++mt) {
#pragma unroll
        for (int r = 0; r < 8; ++r) {
            int m  = mt * 16 + r + hi * 8;
            int mm = (m < NTOK) ? m : (NTOK - 1);   // clamp padded rows (unstored)
            const float* bg = bias_g + ((size_t)mm * NTOK) * HEADS + h;
            float v[4];
#pragma unroll
            for (int nt = 0; nt < 4; ++nt) {
                int n = nt * 16 + cn;
                v[nt] = (n < NTOK) ? accS[mt][nt][r] * scale2 + bg[n * HEADS]
                                   : -3.0e38f;
            }
            float mx = fmaxf(fmaxf(v[0], v[1]), fmaxf(v[2], v[3]));
#pragma unroll
            for (int off = 1; off < 16; off <<= 1)
                mx = fmaxf(mx, __shfl_xor(mx, off, 32));
            float e[4], s = 0.f;
#pragma unroll
            for (int nt = 0; nt < 4; ++nt) {
                e[nt] = ((nt * 16 + cn) < NTOK)
                            ? __builtin_amdgcn_exp2f(v[nt] - mx) : 0.f;
                s += e[nt];
            }
#pragma unroll
            for (int off = 1; off < 16; off <<= 1) s += __shfl_xor(s, off, 32);
            float rs = __builtin_amdgcn_rcpf(s);
#pragma unroll
            for (int nt = 0; nt < 4; ++nt)
                pw[m * NPAD + nt * 16 + cn] = bf16s(e[nt] * rs);
        }
    }

    // O = P @ V : M=64 (4 tiles) x N=32 (2 tiles), K=64 (2 steps)
    v8f accO[4][2];
#pragma unroll
    for (int mt = 0; mt < 4; ++mt)
#pragma unroll
        for (int nt = 0; nt < 2; ++nt) accO[mt][nt] = (v8f){};
#pragma unroll
    for (int ks = 0; ks < 2; ++ks) {
        v16bf aP[4], bV[2];
#pragma unroll
        for (int mt = 0; mt < 4; ++mt)
            aP[mt] = load_a_lds(pw, mt * 16 + cn, NPAD, ks * 32, hi);
#pragma unroll
        for (int nt = 0; nt < 2; ++nt)
            bV[nt] = load_b_lds(vT, h * HD + nt * 16 + cn, NPAD, ks * 32, hi);
#pragma unroll
        for (int mt = 0; mt < 4; ++mt)
#pragma unroll
            for (int nt = 0; nt < 2; ++nt)
                accO[mt][nt] = wmma_bf16(aP[mt], bV[nt], accO[mt][nt]);
    }
#pragma unroll
    for (int mt = 0; mt < 4; ++mt)
#pragma unroll
        for (int nt = 0; nt < 2; ++nt)
#pragma unroll
            for (int r = 0; r < 8; ++r) {
                int m = mt * 16 + r + hi * 8;
                os[m * CDIM + h * HD + nt * 16 + cn] = bf16s(accO[mt][nt][r]);
            }
    __syncthreads();

    // ---------------- phase 3: out = O @ w_proj^T + b_proj -----------------------
    float* og = out + (size_t)b * NTOK * CDIM;
    for (int nt = w * 2; nt < w * 2 + 2; ++nt) {
        int n = nt * 16 + cn;
        v8f acc[4] = {};
#pragma unroll
        for (int ks = 0; ks < 8; ++ks) {
            v16bf bm = load_b_g16(wproj_bf + (size_t)n * CDIM, ks * 32, hi);
#pragma unroll
            for (int mt = 0; mt < 4; ++mt) {
                v16bf a = load_a_lds(os, mt * 16 + cn, CDIM, ks * 32, hi);
                acc[mt] = wmma_bf16(a, bm, acc[mt]);
            }
        }
        float bias = b_proj[n];
#pragma unroll
        for (int mt = 0; mt < 4; ++mt)
#pragma unroll
            for (int r = 0; r < 8; ++r) {
                int m = mt * 16 + r + hi * 8;
                if (m < NTOK) og[m * CDIM + n] = acc[mt][r] + bias;
            }
    }
}

extern "C" void kernel_launch(void* const* d_in, const int* in_sizes, int n_in,
                              void* d_out, int out_size, void* d_ws, size_t ws_size,
                              hipStream_t stream) {
    const float* x          = (const float*)d_in[0];
    const float* w_qkv      = (const float*)d_in[1];
    const float* b_qkv      = (const float*)d_in[2];
    const float* w_proj     = (const float*)d_in[3];
    const float* b_proj     = (const float*)d_in[4];
    const float* bias_table = (const float*)d_in[5];
    float* out = (float*)d_out;

    const int B = in_sizes[0] / (NTOK * CDIM);  // 4096 windows

    // workspace carve: bf16 w_qkv | bf16 w_proj | gathered bias (f32, x log2e)
    unsigned short* wqkv_bf  = (unsigned short*)d_ws;                    // 393216 B
    unsigned short* wproj_bf = (unsigned short*)((char*)d_ws + 393216);  // 131072 B
    float* bias_g = (float*)((char*)d_ws + 524288);                      // 76832 B

    swin_prep<<<dim3((768 * CDIM + 255) / 256), dim3(256), 0, stream>>>(
        w_qkv, w_proj, bias_table, wqkv_bf, wproj_bf, bias_g);

    const size_t smem_bytes =
        (size_t)(4 * NPAD * CDIM + HEADS * NPAD * NPAD) * sizeof(unsigned short);
    (void)hipFuncSetAttribute((const void*)swin_window_attn,
                              hipFuncAttributeMaxDynamicSharedMemorySize,
                              (int)smem_bytes);

    swin_window_attn<<<dim3(B), dim3(256), smem_bytes, stream>>>(
        x, wqkv_bf, b_qkv, wproj_bf, b_proj, bias_g, out);
}
